// CrossNetMix_15229954032206
// MI455X (gfx1250) — matrix operand order repaired and checked by reading, not compiled, not run on hardware
//
#include <hip/hip_runtime.h>

typedef unsigned short u16;
typedef unsigned int u32;
typedef unsigned long long u64;
typedef __attribute__((ext_vector_type(16))) __bf16 v16bf;
typedef __attribute__((ext_vector_type(8))) float v8f;
typedef __attribute__((ext_vector_type(4))) unsigned int tdm_v4u;
typedef __attribute__((ext_vector_type(8))) int tdm_v8i;
typedef __attribute__((ext_vector_type(4))) int tdm_v4i;

#define DIM 512
#define RNK 64
#define NEXP 4
#define NLAYER 3
#define NBATCH 16384
#define P1 520   // k1 A-tile LDS pitch (ushorts): 1024B row + 16B pad
#define P2 264   // h tile LDS pitch (ushorts): 512B row + 16B pad
#define PX 260   // f32 tile LDS pitch: 1024B row + 16B pad

#if __has_builtin(__builtin_amdgcn_tensor_load_to_lds) && \
    __has_builtin(__builtin_amdgcn_s_wait_tensorcnt)
#define USE_TDM 1
#else
#define USE_TDM 0
#endif

__device__ __forceinline__ u16 f32_bf16(float f) {
  u32 u = __float_as_uint(f);
  u32 r = u + 0x7FFFu + ((u >> 16) & 1u);   // round-to-nearest-even
  return (u16)(r >> 16);
}
__device__ __forceinline__ float bf16_f32(u16 h) {
  return __uint_as_float(((u32)h) << 16);
}
// tanh via hardware exp/rcp: exact in saturation tails, ~1e-7 rel error.
__device__ __forceinline__ float fast_tanh(float x) {
  float e = __expf(2.f * x);
  return 1.f - 2.f * __builtin_amdgcn_rcpf(e + 1.f);
}

union FragBF { v16bf v; uint4 q[2]; };

// A-frag: two ds_read_b128 at K offsets {8h..8h+7} and {16+8h..+7} of a chunk.
__device__ __forceinline__ void lda(FragBF& a, const u16* aRow, int chunk) {
  a.q[0] = *(const uint4*)(aRow + 32 * chunk);
  a.q[1] = *(const uint4*)(aRow + 32 * chunk + 16);
}
// B-frag: 32B contiguous per lane in the pre-swizzled weight buffer.
__device__ __forceinline__ void ldb(FragBF& b, const u16* base, int tile,
                                    int chunk, int kc, int lane) {
  const u16* p = base + (size_t)(((tile * kc + chunk) * 32 + lane) * 16);
  b.q[0] = *(const uint4*)p;
  b.q[1] = *(const uint4*)(p + 8);
}
// One K-chunk against 8 consecutive n-tiles, with explicit B double-buffering
// so global_load_b128 of the next fragment overlaps the current v_wmma.
__device__ __forceinline__ void gemm_chunk8(const FragBF& a, const u16* wbase,
                                            int tile0, int chunk, int kc,
                                            int lane, v8f* acc) {
  FragBF b0, b1;
  ldb(b0, wbase, tile0 + 0, chunk, kc, lane);
#pragma unroll
  for (int t = 0; t < 8; t += 2) {
    ldb(b1, wbase, tile0 + t + 1, chunk, kc, lane);
    acc[t] = __builtin_amdgcn_wmma_f32_16x16x32_bf16(
        false, a.v, false, b0.v, (short)0, acc[t], false, false);
    if (t + 2 < 8) ldb(b0, wbase, tile0 + t + 2, chunk, kc, lane);
    acc[t + 1] = __builtin_amdgcn_wmma_f32_16x16x32_bf16(
        false, a.v, false, b1.v, (short)0, acc[t + 1], false, false);
  }
}

#if USE_TDM
// 2-D TDM load: tensor dims == tile dims (no OOB), LDS padding emulates our
// bank-conflict-avoiding pitches. All descriptor fields are wave-uniform.
__device__ __forceinline__ void tdm_load_2d(u32 lds_off, const void* gptr,
                                            u32 dsz_code, u32 tile_d0, u32 tile_d1,
                                            u32 stride_d0, u32 pad_int_code,
                                            u32 pad_amt_code) {
  const u64 ga = (u64)(uintptr_t)gptr;
  tdm_v4u g0;
  g0[0] = 1u;                                 // count=1, user descriptor
  g0[1] = lds_off;                            // lds_addr (bytes)
  g0[2] = (u32)ga;                            // global_addr[31:0]
  g0[3] = (u32)(ga >> 32) | (2u << 30);       // global_addr hi | type=2 (image)
  tdm_v8i g1;
  g1[0] = (int)((dsz_code << 16) | (1u << 20) | (pad_int_code << 22) |
                (pad_amt_code << 25));        // data_size | pad_en | pad codes
  g1[1] = (int)((tile_d0 & 0xFFFFu) << 16);   // tensor_dim0 lo16
  g1[2] = (int)((tile_d0 >> 16) | ((tile_d1 & 0xFFFFu) << 16));  // d0 hi | d1 lo
  g1[3] = (int)((tile_d1 >> 16) | (tile_d0 << 16));              // d1 hi | tile_dim0
  g1[4] = (int)tile_d1;                       // tile_dim1 | tile_dim2=0
  g1[5] = (int)stride_d0;                     // tensor_dim0_stride lo32
  g1[6] = 0;
  g1[7] = 0;
  tdm_v4i gz = {0, 0, 0, 0};
#if __clang_major__ >= 23
  tdm_v8i gz8 = {0, 0, 0, 0, 0, 0, 0, 0};
  __builtin_amdgcn_tensor_load_to_lds(g0, g1, gz, gz, gz8, 0);
#else
  __builtin_amdgcn_tensor_load_to_lds(g0, g1, gz, gz, 0);
#endif
}
#endif

// ---------------------------------------------------------------------------
// Weight prep: f32 -> bf16, swizzled so each lane's 16 B-matrix elements are
// 32 contiguous bytes (two global_load_b128 per fragment).
// B-frag layout: n = 16*t + (lane&15); k = 32*c + 8*(lane>>4) + (j&7) + 16*(j>>3).
// ---------------------------------------------------------------------------
__global__ __launch_bounds__(256) void cnm_prep(
    const float* __restrict__ U, const float* __restrict__ V,
    const float* __restrict__ C,
    u16* __restrict__ Vw, u16* __restrict__ Cw, u16* __restrict__ Uw) {
  const int NV = NLAYER * 16 * 16 * 32;
  const int NU = NLAYER * 32 * 8 * 32;
  const int NC = NLAYER * NEXP * 4 * 2 * 32;
  int gid = blockIdx.x * blockDim.x + threadIdx.x;
  if (gid < NV) {
    const int L = gid & 31; int rest = gid >> 5;
    const int c = rest & 15; rest >>= 4;
    const int t = rest & 15; rest >>= 4;
    const int l = rest;
    const int ln = L & 15, kh = L >> 4;
    const int n = 16 * t + ln;
    const int e = n >> 6, r = n & 63;
    u16* dst = Vw + ((size_t)l * (16 * 16 * 32) + (t * 16 + c) * 32 + L) * 16;
    const float* src = V + (((size_t)l * NEXP + e) * RNK + r) * DIM;
#pragma unroll
    for (int j = 0; j < 16; ++j) dst[j] = f32_bf16(src[32 * c + 8 * kh + (j & 7) + 16 * (j >> 3)]);
  } else if (gid < NV + NU) {
    int g = gid - NV;
    const int L = g & 31; int rest = g >> 5;
    const int c = rest & 7; rest >>= 3;
    const int t = rest & 31; rest >>= 5;
    const int l = rest;
    const int ln = L & 15, kh = L >> 4;
    const int d = 16 * t + ln;
    u16* dst = Uw + ((size_t)l * (32 * 8 * 32) + (t * 8 + c) * 32 + L) * 16;
#pragma unroll
    for (int j = 0; j < 16; ++j) {
      const int k = 32 * c + 8 * kh + (j & 7) + 16 * (j >> 3);
      const int e = k >> 6, r = k & 63;
      dst[j] = f32_bf16(U[(((size_t)l * NEXP + e) * DIM + d) * RNK + r]);
    }
  } else if (gid < NV + NU + NC) {
    int g = gid - NV - NU;
    const int L = g & 31; int rest = g >> 5;
    const int c2 = rest & 1; rest >>= 1;
    const int tl = rest & 3; rest >>= 2;
    const int e = rest & 3; rest >>= 2;
    const int l = rest;
    const int ln = L & 15, kh = L >> 4;
    const int n = 16 * tl + ln;
    u16* dst = Cw + ((size_t)l * (NEXP * 4 * 2 * 32) + ((e * 4 + tl) * 2 + c2) * 32 + L) * 16;
    const float* src = C + (((size_t)l * NEXP + e) * RNK + n) * RNK;
#pragma unroll
    for (int j = 0; j < 16; ++j) dst[j] = f32_bf16(src[32 * c2 + 8 * kh + (j & 7) + 16 * (j >> 3)]);
  }
}

// One-time x (f32) -> bf16 row-major copy for layer 0's k1 staging.
__global__ __launch_bounds__(256) void cnm_xbf(const float* __restrict__ x,
                                               u16* __restrict__ xbf) {
  const size_t i = ((size_t)blockIdx.x * 256 + threadIdx.x) * 8;
  float4 a = *(const float4*)(x + i);
  float4 b = *(const float4*)(x + i + 4);
  union { u16 h[8]; uint4 q; } p;
  p.h[0] = f32_bf16(a.x); p.h[1] = f32_bf16(a.y);
  p.h[2] = f32_bf16(a.z); p.h[3] = f32_bf16(a.w);
  p.h[4] = f32_bf16(b.x); p.h[5] = f32_bf16(b.y);
  p.h[6] = f32_bf16(b.z); p.h[7] = f32_bf16(b.w);
  *(uint4*)(xbf + i) = p.q;
}

// ---------------------------------------------------------------------------
// Kernel 1 (per layer): gate = softmax(xl . gate_w); h1 = tanh(xl V^T);
// h2 = tanh(C h1); h' = gate * h2 (stored bf16, B x 256).
// Workgroup: 256 threads / 8 waves; 64 batch rows; wave tile = 16 x 128.
// ---------------------------------------------------------------------------
__global__ __launch_bounds__(256) void cnm_k1(
    const u16* __restrict__ xbf, const float* __restrict__ gate_w,
    const u16* __restrict__ Vw, const u16* __restrict__ Cw,
    u16* __restrict__ hprime) {
  extern __shared__ char smem_raw[];
  u16* smA = (u16*)smem_raw;                 // 64 x P1 bf16 xl tile
  u16* smH = smA + 64 * P1;                  // 64 x P2 bf16 h1 tile
  float* smGate = (float*)(smH + 64 * P2);   // 64 x 4 gates

  const int tid = threadIdx.x;
  const int row0 = blockIdx.x * 64;

#if USE_TDM
  // TDM: DMA 64x512 bf16 tile into LDS; pad_interval=7 (1024B), pad=3 (16B)
  // reproduces pitch P1 = 520 ushorts.
  if (tid < 32) {
    tdm_load_2d(0u, xbf + (size_t)row0 * DIM, /*dsz 2B*/1u, DIM, 64u, DIM, 7u, 3u);
    __builtin_amdgcn_s_wait_tensorcnt(0);
  }
#else
  {
    const int r = tid >> 2;
    const int c0 = (tid & 3) * 128;
    const u16* src = xbf + (size_t)(row0 + r) * DIM + c0;
    u16* dst = smA + r * P1 + c0;
#pragma unroll
    for (int i = 0; i < 16; ++i)
      *(uint4*)(dst + i * 8) = *(const uint4*)(src + i * 8);
  }
#endif
  __syncthreads();

  // gate logits: one (row, expert) dot product per thread
  {
    const int r = tid >> 2;
    const int e = tid & 3;
    const float* gw = gate_w + e * DIM;
    const u16* xr = smA + r * P1;
    float acc = 0.f;
    for (int k = 0; k < DIM; k += 4) {
      float4 g = *(const float4*)(gw + k);
      acc += bf16_f32(xr[k + 0]) * g.x + bf16_f32(xr[k + 1]) * g.y +
             bf16_f32(xr[k + 2]) * g.z + bf16_f32(xr[k + 3]) * g.w;
    }
    smGate[(r << 2) + e] = acc;
  }
  __syncthreads();
  if (tid < 64) {  // softmax over E=4, in place
    float* g = smGate + (tid << 2);
    float m = fmaxf(fmaxf(g[0], g[1]), fmaxf(g[2], g[3]));
    float e0 = __expf(g[0] - m), e1 = __expf(g[1] - m);
    float e2 = __expf(g[2] - m), e3 = __expf(g[3] - m);
    float inv = __builtin_amdgcn_rcpf(e0 + e1 + e2 + e3);
    g[0] = e0 * inv; g[1] = e1 * inv; g[2] = e2 * inv; g[3] = e3 * inv;
  }
  __syncthreads();

  const int wave = tid >> 5, lane = tid & 31;
  const int wm = wave & 3, nh = wave >> 2;   // m-tile, n-half
  const int ln = lane & 15, kh = lane >> 4;

  // GEMM1: (16 x 512) x (512 x 128) per wave; K-chunk pairs with A prefetch.
  v8f acc[8];
#pragma unroll
  for (int t = 0; t < 8; ++t)
#pragma unroll
    for (int i = 0; i < 8; ++i) acc[t][i] = 0.f;

  const u16* aRow = smA + (16 * wm + ln) * P1 + 8 * kh;
  {
    FragBF aA, aB;
    lda(aA, aRow, 0);
    for (int p = 0; p < 7; ++p) {
      lda(aB, aRow, 2 * p + 1);
      gemm_chunk8(aA, Vw, nh * 8, 2 * p, 16, lane, acc);
      lda(aA, aRow, 2 * p + 2);
      gemm_chunk8(aB, Vw, nh * 8, 2 * p + 1, 16, lane, acc);
    }
    lda(aB, aRow, 15);
    gemm_chunk8(aA, Vw, nh * 8, 14, 16, lane, acc);
    gemm_chunk8(aB, Vw, nh * 8, 15, 16, lane, acc);
  }

  // tanh -> smH; consumed only by this wave (own rows, own expert columns):
  // per-wave LDS ordering makes this barrier-free.
#pragma unroll
  for (int t = 0; t < 8; ++t) {
    const int gt = nh * 8 + t;
#pragma unroll
    for (int v = 0; v < 8; ++v) {
      const int m = 16 * wm + v + 8 * kh;
      smH[m * P2 + gt * 16 + ln] = f32_bf16(fast_tanh(acc[t][v]));
    }
  }

  // GEMM2: per-expert C (K=64, 2 chunks), experts {2*nh, 2*nh+1}
  v8f acc2[8];
#pragma unroll
  for (int t = 0; t < 8; ++t)
#pragma unroll
    for (int i = 0; i < 8; ++i) acc2[t][i] = 0.f;

  const u16* a2Row = smH + (16 * wm + ln) * P2 + 8 * kh;
#pragma unroll
  for (int el = 0; el < 2; ++el) {
    const int e = nh * 2 + el;
#pragma unroll
    for (int c2 = 0; c2 < 2; ++c2) {
      FragBF a;
      a.q[0] = *(const uint4*)(a2Row + e * 64 + 32 * c2);
      a.q[1] = *(const uint4*)(a2Row + e * 64 + 32 * c2 + 16);
      FragBF b0, b1;
      ldb(b0, Cw, e * 4 + 0, c2, 2, lane);
#pragma unroll
      for (int tl = 0; tl < 4; tl += 2) {
        ldb(b1, Cw, e * 4 + tl + 1, c2, 2, lane);
        acc2[el * 4 + tl] = __builtin_amdgcn_wmma_f32_16x16x32_bf16(
            false, a.v, false, b0.v, (short)0, acc2[el * 4 + tl], false, false);
        if (tl + 2 < 4) ldb(b0, Cw, e * 4 + tl + 2, c2, 2, lane);
        acc2[el * 4 + tl + 1] = __builtin_amdgcn_wmma_f32_16x16x32_bf16(
            false, a.v, false, b1.v, (short)0, acc2[el * 4 + tl + 1], false, false);
      }
    }
  }

  // epilogue: tanh, gate scale, store h' as bf16
#pragma unroll
  for (int t = 0; t < 8; ++t) {
    const int gt = nh * 8 + t;
    const int e = gt >> 2;
#pragma unroll
    for (int v = 0; v < 8; ++v) {
      const int m = 16 * wm + v + 8 * kh;
      const float g = smGate[(m << 2) + e];
      hprime[(size_t)(row0 + m) * 256 + gt * 16 + ln] =
          f32_bf16(fast_tanh(acc2[t][v]) * g);
    }
  }
}

// ---------------------------------------------------------------------------
// Kernel 2 (per layer): o = h' Ut (K = E*R = 256, gate already folded in);
// out = xl + x0 * (bias + o); also emits bf16 copy of the new xl for the
// next layer's k1 staging. Workgroup covers 64 rows x 256 cols.
// ---------------------------------------------------------------------------
__global__ __launch_bounds__(256) void cnm_k2(
    const u16* __restrict__ hprime, const u16* __restrict__ Uw,
    const float* __restrict__ bias, const float* __restrict__ xl,
    const float* __restrict__ x0, float* __restrict__ out,
    u16* __restrict__ xbf_out) {
  extern __shared__ char smem_raw[];
  u16* smH = (u16*)smem_raw;                 // 64 x P2 bf16 h' tile
  float* smXL = (float*)(smH + 64 * P2);     // 64 x PX f32
  float* smX0 = smXL + 64 * PX;              // 64 x PX f32

  const int tid = threadIdx.x;
  const int row0 = blockIdx.x * 64;
  const int col0 = blockIdx.y * 256;

#if USE_TDM
  if (tid < 32) {
    // h': 512B rows + 16B pad -> pitch P2; xl/x0: 1024B rows + 16B pad -> PX
    tdm_load_2d(0u, hprime + (size_t)row0 * 256, /*2B*/1u, 256u, 64u, 256u, 6u, 3u);
    tdm_load_2d((u32)(64 * P2 * 2), xl + (size_t)row0 * DIM + col0,
                /*4B*/2u, 256u, 64u, DIM, 7u, 3u);
    tdm_load_2d((u32)(64 * P2 * 2 + 64 * PX * 4), x0 + (size_t)row0 * DIM + col0,
                /*4B*/2u, 256u, 64u, DIM, 7u, 3u);
    __builtin_amdgcn_s_wait_tensorcnt(0);
  }
#else
  {
    const int r = tid >> 2;
    const int c0 = (tid & 3) * 64;
    const u16* src = hprime + (size_t)(row0 + r) * 256 + c0;
    u16* dst = smH + r * P2 + c0;
#pragma unroll
    for (int i = 0; i < 8; ++i)
      *(uint4*)(dst + i * 8) = *(const uint4*)(src + i * 8);
    const float* s1 = xl + (size_t)(row0 + r) * DIM + col0 + c0;
    const float* s2 = x0 + (size_t)(row0 + r) * DIM + col0 + c0;
    float* d1 = smXL + r * PX + c0;
    float* d2 = smX0 + r * PX + c0;
#pragma unroll
    for (int i = 0; i < 16; ++i) {
      *(float4*)(d1 + i * 4) = *(const float4*)(s1 + i * 4);
      *(float4*)(d2 + i * 4) = *(const float4*)(s2 + i * 4);
    }
  }
#endif
  __syncthreads();

  const int wave = tid >> 5, lane = tid & 31;
  const int wm = wave & 3, nh = wave >> 2;
  const int ln = lane & 15, kh = lane >> 4;

  v8f acc[8];
#pragma unroll
  for (int t = 0; t < 8; ++t)
#pragma unroll
    for (int i = 0; i < 8; ++i) acc[t][i] = 0.f;

  const u16* aRow = smH + (16 * wm + ln) * P2 + 8 * kh;
  const int tile0 = (col0 >> 4) + nh * 8;  // global n-tile base over D
  {
    FragBF aA, aB;
    lda(aA, aRow, 0);
    for (int p = 0; p < 3; ++p) {
      lda(aB, aRow, 2 * p + 1);
      gemm_chunk8(aA, Uw, tile0, 2 * p, 8, lane, acc);
      lda(aA, aRow, 2 * p + 2);
      gemm_chunk8(aB, Uw, tile0, 2 * p + 1, 8, lane, acc);
    }
    lda(aB, aRow, 7);
    gemm_chunk8(aA, Uw, tile0, 6, 8, lane, acc);
    gemm_chunk8(aB, Uw, tile0, 7, 8, lane, acc);
  }

#pragma unroll
  for (int t = 0; t < 8; ++t) {
    const int nl = (nh * 8 + t) * 16 + ln;  // 0..255 within this column group
    const int dcol = col0 + nl;
    const float bs = bias[dcol];
#pragma unroll
    for (int v = 0; v < 8; ++v) {
      const int m = 16 * wm + v + 8 * kh;
      const float val = smXL[m * PX + nl] + smX0[m * PX + nl] * (acc[t][v] + bs);
      out[(size_t)(row0 + m) * DIM + dcol] = val;
      xbf_out[(size_t)(row0 + m) * DIM + dcol] = f32_bf16(val);
    }
  }
}

// ---------------------------------------------------------------------------
// Launch: prep weights + x->bf16 once, then 3 layers x (k1, k2).
// f32 xl ping-pong: x -> d_out -> ws.bufA -> d_out; bf16 xl lives in ws.xbf.
// ---------------------------------------------------------------------------
extern "C" void kernel_launch(void* const* d_in, const int* in_sizes, int n_in,
                              void* d_out, int out_size, void* d_ws, size_t ws_size,
                              hipStream_t stream) {
  (void)in_sizes; (void)n_in; (void)out_size; (void)ws_size;
  const float* x    = (const float*)d_in[0];
  const float* U    = (const float*)d_in[1];
  const float* V    = (const float*)d_in[2];
  const float* C    = (const float*)d_in[3];
  const float* bias = (const float*)d_in[4];
  const float* gw   = (const float*)d_in[5];
  float* out = (float*)d_out;

  char* ws = (char*)d_ws;
  u16* Vw   = (u16*)(ws);                              //   786432 B
  u16* Cw   = (u16*)(ws + 786432);                     //    98304 B
  u16* Uw   = (u16*)(ws + 786432 + 98304);             //   786432 B
  u16* hp   = (u16*)(ws + 1671168);                    //  8388608 B
  float* bufA = (float*)(ws + 1671168 + 8388608);      // 33554432 B
  u16* xbf  = (u16*)(ws + 1671168 + 8388608 + 33554432); // 16777216 B (~57.6 MB)

  {
    const int total = NLAYER * 16 * 16 * 32 + NLAYER * 32 * 8 * 32 +
                      NLAYER * NEXP * 4 * 2 * 32;
    cnm_prep<<<(total + 255) / 256, 256, 0, stream>>>(U, V, C, Vw, Cw, Uw);
    cnm_xbf<<<(NBATCH * DIM / 8 + 255) / 256, 256, 0, stream>>>(x, xbf);
  }

  const size_t shm1 = (size_t)(64 * P1 + 64 * P2) * 2 + 256 * 4;      // ~101 KB
  const size_t shm2 = (size_t)64 * P2 * 2 + (size_t)2 * 64 * PX * 4;  // ~163 KB

  for (int l = 0; l < NLAYER; ++l) {
    const float* xin = (l == 0) ? x : (l == 1) ? out : bufA;
    float* xout = (l == 1) ? bufA : out;
    cnm_k1<<<dim3(NBATCH / 64), 256, shm1, stream>>>(
        xbf, gw, Vw + (size_t)l * 16 * 16 * 32 * 16,
        Cw + (size_t)l * 1024 * 16, hp);
    cnm_k2<<<dim3(NBATCH / 64, 2), 256, shm2, stream>>>(
        hp, Uw + (size_t)l * 32 * 8 * 32 * 16, bias + (size_t)l * DIM,
        xin, x, xout, xbf);
  }
}
